// KalmanNetNN_21818433863784
// MI455X (gfx1250) — matrix operand order repaired and checked by reading, not compiled, not run on hardware
//
#include <hip/hip_runtime.h>

typedef __attribute__((ext_vector_type(16))) __bf16 v16bf;
typedef __attribute__((ext_vector_type(8)))  float  v8f;

#define NWG   72
#define TPB   256
#define NWAVE (NWG * (TPB / 32))

// Problem sizes (hardcoded per reference)
#define TT   512
#define H1   5120
#define H2   4096
#define HID  1000
#define HIDP 1024
#define G3   3000   // 3*HID
#define G3P  3008   // padded to 16

// Tile counts: rows/16, K-chunks of 32
#define T_W1  320
#define KC_W1   2
#define T_WIH 188
#define KC_WIH 160
#define KHALF   80   // K-split of Wih tiles (2 partials)
#define T_WHH 188
#define KC_WHH  32
#define T_W2  256
#define KC_W2   32
#define T_W3   64
#define KC_W3  128
#define KQUAD   32   // K-split of W3 tiles (4 partials)

#define JOBS_B (2 * T_WIH + T_WHH)   // 376 gi-halves + 188 gh = 564
#define JOBS_E (4 * T_W3)            // 256

#if __has_builtin(__builtin_amdgcn_global_load_async_to_lds_b128) && \
    __has_builtin(__builtin_amdgcn_s_wait_asynccnt)
#define HAS_ASYNC_LDS 1
#else
#define HAS_ASYNC_LDS 0
#endif

// ---------------------------------------------------------------------------
// Pack fp32 row-major weights -> bf16 in WMMA A-fragment order.
// Fragment for (tile, kc, lane) stored contiguously at ((tile*kcn+kc)*32+lane)*16.
// contiguous-k A 16x32 packing used here (B matches): lanes 0-15: K = kc*32 + {0..15}
//                        lanes 16-31: K = kc*32 + {16..31}, M = tile*16 + (lane&15)
// ---------------------------------------------------------------------------
__global__ void pack_bf16(const float* __restrict__ src, __bf16* __restrict__ dst,
                          int rows, int cols, int tiles, int kcn) {
  int idx = blockIdx.x * blockDim.x + threadIdx.x;
  int total = tiles * kcn * 32;
  if (idx >= total) return;
  int lane = idx & 31;
  int rest = idx >> 5;
  int kc   = rest % kcn;
  int tile = rest / kcn;
  int m = tile * 16 + (lane & 15);
  int kbase = kc * 32 + ((lane < 16) ? 0 : 16);
  __bf16* d = dst + (size_t)idx * 16;
#pragma unroll
  for (int e = 0; e < 16; ++e) {
    int k = kbase + e;
    float v = (m < rows && k < cols) ? src[(size_t)m * cols + k] : 0.0f;
    d[e] = (__bf16)v;
  }
}

__global__ void init_counters(unsigned* cnt) {
  cnt[0] = 0u;  // arrive
  cnt[1] = 0u;  // generation
}

// ---------------------------------------------------------------------------
// Grid-wide barrier (persistent kernel, NWG resident workgroups)
// ---------------------------------------------------------------------------
__device__ __forceinline__ void grid_sync(unsigned* cnt, unsigned& gen) {
  __threadfence();
  __syncthreads();
  if (threadIdx.x == 0) {
    unsigned prev = __hip_atomic_fetch_add(&cnt[0], 1u, __ATOMIC_ACQ_REL,
                                           __HIP_MEMORY_SCOPE_AGENT);
    if (prev == NWG - 1) {
      __hip_atomic_store(&cnt[0], 0u, __ATOMIC_RELAXED, __HIP_MEMORY_SCOPE_AGENT);
      __hip_atomic_fetch_add(&cnt[1], 1u, __ATOMIC_RELEASE, __HIP_MEMORY_SCOPE_AGENT);
    } else {
      while (__hip_atomic_load(&cnt[1], __ATOMIC_ACQUIRE, __HIP_MEMORY_SCOPE_AGENT) <= gen) {
        __builtin_amdgcn_s_sleep(2);
      }
    }
    gen++;
  }
  __syncthreads();
  __threadfence();
}

// ---------------------------------------------------------------------------
// Stage a global bf16 buffer into LDS. Uses gfx1250 async global->LDS DMA
// (ASYNCcnt-tracked, bypasses VGPRs) when the builtin exists.
// Builtin signature (from compiler diagnostic): param0 is
// `__attribute__((vector_size(16))) int` pointer in global address space.
// ---------------------------------------------------------------------------
__device__ __forceinline__ void stage_to_lds(void* lds_dst, const void* gsrc, int bytes) {
  const int tid = threadIdx.x;
#if HAS_ASYNC_LDS
  typedef __attribute__((__vector_size__(16))) int v4i_;
  typedef __attribute__((address_space(1))) v4i_ gvec_t;
  typedef __attribute__((address_space(3))) v4i_ lvec_t;
  gvec_t* g = (gvec_t*)(void*)gsrc;
  lvec_t* l = (lvec_t*)lds_dst;
  for (int i = tid; i < bytes / 16; i += TPB) {
    __builtin_amdgcn_global_load_async_to_lds_b128(g + i, l + i, 0, 0);
  }
  __builtin_amdgcn_s_wait_asynccnt(0);
#else
  const uint4* src = (const uint4*)gsrc;
  uint4* dst = (uint4*)lds_dst;
  for (int i = tid; i < bytes / 16; i += TPB) dst[i] = src[i];
#endif
  __syncthreads();
}

// ---------------------------------------------------------------------------
// 16-row mat-vec over nk K-chunks via WMMA bf16, PF-deep software pipeline.
// wfrag already includes tile/kc0 base; vec already includes kc0*32.
// D: lane 0 holds rows tile*16+0..7 in c[0..7]; lane 16 rows +8..15.
// Requires nk % PF == 0.
// ---------------------------------------------------------------------------
template <int PF>
__device__ __forceinline__ v8f matvec_range(const __bf16* __restrict__ wfrag,
                                            const __bf16* __restrict__ vec, int nk) {
  const int lane = threadIdx.x & 31;
  const __bf16* ap = wfrag + lane * 16;
  const __bf16* bp = vec + ((lane & 16) ? 16 : 0);
  v16bf a[PF], b[PF];
#pragma unroll
  for (int i = 0; i < PF; ++i) {
    a[i] = *(const v16bf*)(ap + (size_t)i * 512);
    b[i] = *(const v16bf*)(bp + (size_t)i * 32);
  }
  v8f c = {0.f, 0.f, 0.f, 0.f, 0.f, 0.f, 0.f, 0.f};
  const int groups = nk / PF;
#pragma unroll 1
  for (int g = 1; g < groups; ++g) {
#pragma unroll
    for (int i = 0; i < PF; ++i) {
      const int kc = g * PF + i;
      v16bf an = *(const v16bf*)(ap + (size_t)kc * 512);
      v16bf bn = *(const v16bf*)(bp + (size_t)kc * 32);
      c = __builtin_amdgcn_wmma_f32_16x16x32_bf16(false, a[i], false, b[i],
                                                  (short)0, c, false, false);
      a[i] = an; b[i] = bn;
    }
  }
#pragma unroll
  for (int i = 0; i < PF; ++i)
    c = __builtin_amdgcn_wmma_f32_16x16x32_bf16(false, a[i], false, b[i],
                                                (short)0, c, false, false);
  return c;
}

__device__ __forceinline__ float sigmoidf_(float x) {
  return 1.0f / (1.0f + __expf(-x));
}

// ---------------------------------------------------------------------------
// Persistent fused KalmanNet kernel
// ---------------------------------------------------------------------------
__global__ __launch_bounds__(TPB, 1) void kalman_persistent(
    const float* __restrict__ y, const float* __restrict__ F,
    const float* __restrict__ Hm, const float* __restrict__ m1x0,
    const float* __restrict__ h0, const float* __restrict__ b1,
    const float* __restrict__ bih, const float* __restrict__ bhh,
    const float* __restrict__ b2, const float* __restrict__ b3,
    const __bf16* __restrict__ W1P, const __bf16* __restrict__ WihP,
    const __bf16* __restrict__ WhhP, const __bf16* __restrict__ W2P,
    const __bf16* __restrict__ W3P,
    __bf16* __restrict__ xb, float* __restrict__ giA, float* __restrict__ giB,
    float* __restrict__ gh, __bf16* __restrict__ l2b, float* __restrict__ kgp,
    unsigned* cnt, float* __restrict__ out) {
  __shared__ float s_post[32], s_prior[32], s_pnew[32], s_m1y[32];
  __shared__ float s_dy[32], s_dmx[32], s_nrm[2];
  __shared__ float s_h[HID];
  __shared__ __attribute__((aligned(16))) __bf16 s_hb[HIDP];
  __shared__ __attribute__((aligned(16))) __bf16 s_kin[64];
  __shared__ __attribute__((aligned(16))) __bf16 s_xb[H1];
  __shared__ __attribute__((aligned(16))) __bf16 s_l2b[H2];

  const int tid  = threadIdx.x;
  const int lane = tid & 31;
  const int gwave = blockIdx.x * (TPB / 32) + (tid >> 5);
  unsigned gen = 0;

  // --- init per-WG redundant state ---
  if (tid < 32) { s_post[tid] = m1x0[tid]; s_prior[tid] = m1x0[tid]; }
  for (int e = tid; e < HID; e += TPB) { s_h[e] = h0[e]; s_hb[e] = (__bf16)h0[e]; }
  for (int e = HID + tid; e < HIDP; e += TPB) s_hb[e] = (__bf16)0.0f;
  __syncthreads();

  for (int t = 0; t < TT; ++t) {
    // ---- L0: Kalman small algebra + feature vector (redundant per WG) ----
    if (tid < 32) {
      float acc = 0.f;
      for (int j = 0; j < 32; ++j) acc += F[tid * 32 + j] * s_post[j];
      s_pnew[tid] = acc;                         // m1x_prior = F @ m1x_post
    }
    __syncthreads();
    if (tid < 32) {
      float acc = 0.f;
      for (int j = 0; j < 32; ++j) acc += Hm[tid * 32 + j] * s_pnew[j];
      s_m1y[tid] = acc;                          // m1y = H @ m1x_prior
    }
    __syncthreads();
    if (tid < 32) {
      s_dy[tid]  = y[t * 32 + tid] - s_m1y[tid]; // innovation
      s_dmx[tid] = s_post[tid] - s_prior[tid];   // post - prev_prior
    }
    __syncthreads();
    if (tid == 0) {
      float a = 0.f, b = 0.f;
      for (int i = 0; i < 32; ++i) { a += s_dy[i] * s_dy[i]; b += s_dmx[i] * s_dmx[i]; }
      s_nrm[0] = fmaxf(__fsqrt_rn(a), 1e-12f);
      s_nrm[1] = fmaxf(__fsqrt_rn(b), 1e-12f);
    }
    __syncthreads();
    if (tid < 32) {
      s_kin[tid]      = (__bf16)(s_dy[tid]  / s_nrm[0]);
      s_kin[32 + tid] = (__bf16)(s_dmx[tid] / s_nrm[1]);
      s_prior[tid] = s_pnew[tid];                // carry new prior
    }
    __syncthreads();

    // ---- Stage A: x = relu(W1 @ kin + b1) -> xb (bf16, global) ----
    for (int job = gwave; job < T_W1; job += NWAVE) {
      v8f c = matvec_range<2>(W1P + (size_t)job * KC_W1 * 512, s_kin, KC_W1);
      if ((lane & 15) == 0) {
        int row0 = job * 16 + ((lane >> 4) << 3);
#pragma unroll
        for (int v = 0; v < 8; ++v) {
          int r = row0 + v;
          xb[r] = (__bf16)fmaxf(c[v] + b1[r], 0.0f);
        }
      }
    }
    grid_sync(cnt, gen);

    // stage x into LDS (async DMA on CDNA5), then broadcast from LDS banks
    stage_to_lds(s_xb, xb, H1 * 2);

    // ---- Stage B: gi = Wih @ x + bih (K-split x2) ; gh = Whh @ h + bhh ----
    for (int job = gwave; job < JOBS_B; job += NWAVE) {
      if (job < 2 * T_WIH) {
        int tile = job >> 1, half = job & 1;
        v8f c = matvec_range<8>(WihP + ((size_t)tile * KC_WIH + half * KHALF) * 512,
                                s_xb + half * KHALF * 32, KHALF);
        float* dstb = half ? giB : giA;
        if ((lane & 15) == 0) {
          int row0 = tile * 16 + ((lane >> 4) << 3);
#pragma unroll
          for (int v = 0; v < 8; ++v) {
            int r = row0 + v;
            if (r < G3) dstb[r] = c[v] + (half ? 0.0f : bih[r]);
          }
        }
      } else {
        int tl = job - 2 * T_WIH;
        v8f c = matvec_range<8>(WhhP + (size_t)tl * KC_WHH * 512, s_hb, KC_WHH);
        if ((lane & 15) == 0) {
          int row0 = tl * 16 + ((lane >> 4) << 3);
#pragma unroll
          for (int v = 0; v < 8; ++v) {
            int r = row0 + v;
            if (r < G3) gh[r] = c[v] + bhh[r];
          }
        }
      }
    }
    grid_sync(cnt, gen);

    // ---- L1: GRU gates (redundant per WG; h kept purely in LDS) ----
    for (int e = tid; e < HID; e += TPB) {
      float ir = giA[e] + giB[e];
      float iz = giA[HID + e] + giB[HID + e];
      float ic = giA[2 * HID + e] + giB[2 * HID + e];
      float hr = gh[e], hz = gh[HID + e], hc = gh[2 * HID + e];
      float r  = sigmoidf_(ir + hr);
      float z  = sigmoidf_(iz + hz);
      float cc = tanhf(ic + r * hc);
      float hn = (1.0f - z) * cc + z * s_h[e];
      s_h[e]  = hn;
      s_hb[e] = (__bf16)hn;
    }
    __syncthreads();

    // ---- Stage D: l2 = relu(W2 @ h_new + b2) -> l2b (bf16, global) ----
    for (int job = gwave; job < T_W2; job += NWAVE) {
      v8f c = matvec_range<8>(W2P + (size_t)job * KC_W2 * 512, s_hb, KC_W2);
      if ((lane & 15) == 0) {
        int row0 = job * 16 + ((lane >> 4) << 3);
#pragma unroll
        for (int v = 0; v < 8; ++v) {
          int r = row0 + v;
          l2b[r] = (__bf16)fmaxf(c[v] + b2[r], 0.0f);
        }
      }
    }
    grid_sync(cnt, gen);

    // stage l2 into LDS (async DMA on CDNA5)
    stage_to_lds(s_l2b, l2b, H2 * 2);

    // ---- Stage E: KG = W3 @ l2 + b3 (K-split x4 -> kgp partials) ----
    for (int job = gwave; job < JOBS_E; job += NWAVE) {
      int tile = job >> 2, q = job & 3;
      v8f c = matvec_range<8>(W3P + ((size_t)tile * KC_W3 + q * KQUAD) * 512,
                              s_l2b + q * KQUAD * 32, KQUAD);
      if ((lane & 15) == 0) {
        int row0 = tile * 16 + ((lane >> 4) << 3);
        float* dstk = kgp + q * 1024;
#pragma unroll
        for (int v = 0; v < 8; ++v) {
          int r = row0 + v;
          dstk[r] = c[v] + ((q == 0) ? b3[r] : 0.0f);
        }
      }
    }
    grid_sync(cnt, gen);

    // ---- L2: innovation update (redundant per WG); WG0 writes output ----
    // (no trailing grid barrier needed: next step's stage-A barrier orders
    //  every cross-WG buffer before it is overwritten)
    if (tid < 32) {
      float acc = s_prior[tid];                  // = m1x_prior (pnew)
      for (int j = 0; j < 32; ++j) {
        int idx = tid * 32 + j;
        float k = kgp[idx] + kgp[1024 + idx] + kgp[2048 + idx] + kgp[3072 + idx];
        acc += k * s_dy[j];
      }
      s_post[tid] = acc;
      if (blockIdx.x == 0) out[t * 32 + tid] = acc;
    }
    __syncthreads();
  }
}

// ---------------------------------------------------------------------------
extern "C" void kernel_launch(void* const* d_in, const int* in_sizes, int n_in,
                              void* d_out, int out_size, void* d_ws, size_t ws_size,
                              hipStream_t stream) {
  const float* y    = (const float*)d_in[0];
  const float* F    = (const float*)d_in[1];
  const float* Hm   = (const float*)d_in[2];
  const float* m1x0 = (const float*)d_in[3];
  const float* h0   = (const float*)d_in[4];
  const float* W1   = (const float*)d_in[5];
  const float* b1   = (const float*)d_in[6];
  const float* Wih  = (const float*)d_in[7];
  const float* bih  = (const float*)d_in[8];
  const float* Whh  = (const float*)d_in[9];
  const float* bhh  = (const float*)d_in[10];
  const float* W2   = (const float*)d_in[11];
  const float* b2   = (const float*)d_in[12];
  const float* W3   = (const float*)d_in[13];
  const float* b3   = (const float*)d_in[14];
  float* out = (float*)d_out;

  char* ws = (char*)d_ws;
  size_t off = 0;
  auto alloc = [&](size_t bytes) -> char* {
    char* p = ws + off;
    off += bytes;
    off = (off + 255) & ~(size_t)255;
    return p;
  };
  unsigned* cnt = (unsigned*)alloc(256);
  __bf16* xb   = (__bf16*)alloc((size_t)H1 * 2);
  float*  giA  = (float*) alloc((size_t)G3P * 4);
  float*  giB  = (float*) alloc((size_t)G3P * 4);
  float*  gh   = (float*) alloc((size_t)G3P * 4);
  __bf16* l2b  = (__bf16*)alloc((size_t)H2 * 2);
  float*  kgp  = (float*) alloc((size_t)4 * 1024 * 4);
  __bf16* W1P  = (__bf16*)alloc((size_t)T_W1  * KC_W1  * 512 * 2);
  __bf16* WihP = (__bf16*)alloc((size_t)T_WIH * KC_WIH * 512 * 2);
  __bf16* WhhP = (__bf16*)alloc((size_t)T_WHH * KC_WHH * 512 * 2);
  __bf16* W2P  = (__bf16*)alloc((size_t)T_W2  * KC_W2  * 512 * 2);
  __bf16* W3P  = (__bf16*)alloc((size_t)T_W3  * KC_W3  * 512 * 2);
  (void)ws_size; (void)in_sizes; (void)n_in; (void)out_size;

  hipLaunchKernelGGL(init_counters, dim3(1), dim3(1), 0, stream, cnt);

  auto pack = [&](const float* src, __bf16* dst, int rows, int cols, int tiles, int kcn) {
    int total = tiles * kcn * 32;
    hipLaunchKernelGGL(pack_bf16, dim3((total + TPB - 1) / TPB), dim3(TPB), 0, stream,
                       src, dst, rows, cols, tiles, kcn);
  };
  pack(W1,  W1P,  H1,  64,   T_W1,  KC_W1);
  pack(Wih, WihP, G3,  H1,   T_WIH, KC_WIH);
  pack(Whh, WhhP, G3,  HID,  T_WHH, KC_WHH);
  pack(W2,  W2P,  H2,  HID,  T_W2,  KC_W2);
  pack(W3,  W3P,  1024, H2,  T_W3,  KC_W3);

  hipLaunchKernelGGL(kalman_persistent, dim3(NWG), dim3(TPB), 0, stream,
                     y, F, Hm, m1x0, h0, b1, bih, bhh, b2, b3,
                     W1P, WihP, WhhP, W2P, W3P,
                     xb, giA, giB, gh, l2b, kgp, cnt, out);
}
